// Attention_326417515190
// MI455X (gfx1250) — compile-verified, matrix operand order checked by reading, not run on hardware
//
#include <hip/hip_runtime.h>

typedef _Float16 v2h  __attribute__((ext_vector_type(2)));
typedef _Float16 v4h  __attribute__((ext_vector_type(4)));
typedef _Float16 v8h  __attribute__((ext_vector_type(8)));
typedef _Float16 v16h __attribute__((ext_vector_type(16)));
typedef float    v8f  __attribute__((ext_vector_type(8)));

#define HEADS 16
#define DHEAD 64
#define SEQ   2048
#define DIM   1024
#define BATCH 2
#define MROWS (BATCH*SEQ)                       /* 4096 */
#define N_QKV (3*DIM)                           /* 3072 */
#define HSZ   ((size_t)SEQ*DHEAD)               /* 131072 elems per head */
#define QKV_ELEMS ((size_t)BATCH*HEADS*SEQ*DHEAD) /* 4194304 */

__device__ __forceinline__ v8f v8f_zero() {
  v8f z;
#pragma unroll
  for (int i = 0; i < 8; ++i) z[i] = 0.0f;
  return z;
}

// Load a 16-lane-striped A/B fragment: lane r holds row r; K values are two
// contiguous 8-element chunks at K = half*8 and K = 16 + half*8 (ISA 7.12.2).
__device__ __forceinline__ v16h frag_ld(const _Float16* rowp, int half) {
  const v8h lo = *reinterpret_cast<const v8h*>(rowp + half * 8);
  const v8h hi = *reinterpret_cast<const v8h*>(rowp + 16 + half * 8);
  v16h a;
#pragma unroll
  for (int i = 0; i < 8; ++i) { a[i] = lo[i]; a[i + 8] = hi[i]; }
  return a;
}

__device__ __forceinline__ v8f wmma16x16x32(v16h a, v16h b, v8f c) {
  return __builtin_amdgcn_wmma_f32_16x16x32_f16(false, a, false, b, (short)0, c,
                                                false, false);
}

// Async copy 16 bytes global -> LDS (gfx1250 GLOBAL_LOAD_ASYNC_TO_LDS_B128,
// tracked with ASYNCcnt; no VGPR data movement).  ldsaddr = low 32 bits of the
// generic shared pointer (LDS aperture offset == wave-relative LDS address).
__device__ __forceinline__ void async_copy_b128(const void* src, void* lds_dst) {
  const unsigned ldsaddr = (unsigned)(size_t)lds_dst;
  asm volatile("global_load_async_to_lds_b128 %0, %1, off offset:0"
               :: "v"(ldsaddr), "v"(src) : "memory");
}
__device__ __forceinline__ void async_wait_all() {
  asm volatile("s_wait_asynccnt 0" ::: "memory");
}

// ---------------------------------------------------------------------------
// Kernel 1: QKV projection.  C[4096,3072] = X[4096,1024] * Wqkv[1024,3072]
// fp32 in, f16 out scattered head-major: Q[bh][n][d], K[bh][n][d], Vt[bh][d][n]
// ---------------------------------------------------------------------------
__global__ __launch_bounds__(256)
void qkv_gemm_kernel(const float* __restrict__ X, const float* __restrict__ Wqkv,
                     _Float16* __restrict__ Qw, _Float16* __restrict__ Kw,
                     _Float16* __restrict__ Vt) {
  __shared__ __align__(16) _Float16 As[128][32];
  __shared__ __align__(16) _Float16 Bs[128][32];   // stored transposed: [n][k]
  const int tid  = threadIdx.x;
  const int m0   = blockIdx.y * 128;
  const int n0   = blockIdx.x * 128;
  const int wave = tid >> 5, lane = tid & 31;
  const int r    = lane & 15, half = lane >> 4;
  const int wm   = wave & 3,  wn   = wave >> 2;

  v8f acc[2][4];
#pragma unroll
  for (int t = 0; t < 2; ++t)
#pragma unroll
    for (int u = 0; u < 4; ++u) acc[t][u] = v8f_zero();

  for (int kt = 0; kt < DIM / 32; ++kt) {
    const int k0 = kt * 32;
#pragma unroll
    for (int i = 0; i < 4; ++i) {          // A tile: 128x32 fp32 -> f16 (packed)
      const int lin = tid + i * 256;
      const int row = lin >> 3, c4 = (lin & 7) * 4;
      const float4 v = *reinterpret_cast<const float4*>(
          X + (size_t)(m0 + row) * DIM + k0 + c4);
      const v4h p = {(_Float16)v.x, (_Float16)v.y, (_Float16)v.z, (_Float16)v.w};
      *reinterpret_cast<v4h*>(&As[row][c4]) = p;   // one ds_store_b64
    }
#pragma unroll
    for (int i = 0; i < 4; ++i) {          // B tile: 32x128 fp32 -> f16, transposed
      const int lin  = tid + i * 256;
      const int krow = lin >> 5, c4 = (lin & 31) * 4;
      const float4 v = *reinterpret_cast<const float4*>(
          Wqkv + (size_t)(k0 + krow) * N_QKV + n0 + c4);
      Bs[c4 + 0][krow] = (_Float16)v.x;
      Bs[c4 + 1][krow] = (_Float16)v.y;
      Bs[c4 + 2][krow] = (_Float16)v.z;
      Bs[c4 + 3][krow] = (_Float16)v.w;
    }
    __syncthreads();
    v16h af[2], bf[4];
#pragma unroll
    for (int t = 0; t < 2; ++t) af[t] = frag_ld(&As[wm * 32 + t * 16 + r][0], half);
#pragma unroll
    for (int u = 0; u < 4; ++u) bf[u] = frag_ld(&Bs[wn * 64 + u * 16 + r][0], half);
#pragma unroll
    for (int t = 0; t < 2; ++t)
#pragma unroll
      for (int u = 0; u < 4; ++u) acc[t][u] = wmma16x16x32(af[t], bf[u], acc[t][u]);
    __syncthreads();
  }

  // Epilogue: scatter f16 results into head-major Q / K / transposed V.
#pragma unroll
  for (int t = 0; t < 2; ++t)
#pragma unroll
    for (int u = 0; u < 4; ++u) {
      const int c     = n0 + wn * 64 + u * 16 + r;
      const int which = c >> 10;              // 0=Q 1=K 2=V (tiles never straddle)
      const int cw    = c & 1023;
      const int h     = cw >> 6, d = cw & 63;
#pragma unroll
      for (int e = 0; e < 8; ++e) {
        const int m = m0 + wm * 32 + t * 16 + e + 8 * half;
        const int b = m >> 11, n = m & 2047;
        const _Float16 val  = (_Float16)acc[t][u][e];
        const size_t   head = (size_t)(b * HEADS + h);
        if (which == 0)      Qw[(head * SEQ + n) * DHEAD + d] = val;
        else if (which == 1) Kw[(head * SEQ + n) * DHEAD + d] = val;
        else                 Vt[(head * DHEAD + d) * SEQ + n] = val;
      }
    }
}

// ---------------------------------------------------------------------------
// Kernel 2: fused (flash) attention.  One wave owns a 16-row query tile;
// block = 4 waves = 64 query rows.  Streams 64-key blocks with online softmax.
// ---------------------------------------------------------------------------
__global__ __launch_bounds__(128)
void attn_kernel(const _Float16* __restrict__ Qw, const _Float16* __restrict__ Kw,
                 const _Float16* __restrict__ Vt, _Float16* __restrict__ AO) {
  __shared__ __align__(16) _Float16 Plds[4][16][64];   // per-wave P staging
  const int tid  = threadIdx.x;
  const int wave = tid >> 5, lane = tid & 31;
  const int r    = lane & 15, half = lane >> 4;
  const int bh   = blockIdx.y;
  const int q0   = blockIdx.x * 64 + wave * 16;

  const _Float16* Qh = Qw + (size_t)bh * HSZ;
  const _Float16* Kh = Kw + (size_t)bh * HSZ;
  const _Float16* Vh = Vt + (size_t)bh * HSZ;   // [d][n] layout

  const float kscale = 0.125f * 1.44269504088896340736f;  // SCALE * log2(e)

  v16h qa[2];
#pragma unroll
  for (int ks = 0; ks < 2; ++ks)
    qa[ks] = frag_ld(Qh + (size_t)(q0 + r) * DHEAD + ks * 32, half);

  v8f o[4];
#pragma unroll
  for (int nt = 0; nt < 4; ++nt) o[nt] = v8f_zero();
  float mrow[8], lrow[8];
#pragma unroll
  for (int e = 0; e < 8; ++e) { mrow[e] = -3.0e38f; lrow[e] = 0.0f; }

  for (int kb = 0; kb < SEQ / 64; ++kb) {
    const int kbase = kb * 64;
    if (kb + 1 < SEQ / 64) {                 // hint next K block toward cache
      __builtin_prefetch(Kh + (size_t)(kbase + 64 + r) * DHEAD, 0, 1);
      __builtin_prefetch(Vh + (size_t)r * SEQ + kbase + 64, 0, 1);
    }
    // S = Q * K^T (K rows load with the same A-style fragment pattern)
    v8f s[4];
#pragma unroll
    for (int nt = 0; nt < 4; ++nt) s[nt] = v8f_zero();
#pragma unroll
    for (int ks = 0; ks < 2; ++ks)
#pragma unroll
      for (int nt = 0; nt < 4; ++nt) {
        const v16h kf =
            frag_ld(Kh + (size_t)(kbase + nt * 16 + r) * DHEAD + ks * 32, half);
        s[nt] = wmma16x16x32(qa[ks], kf, s[nt]);
      }
#pragma unroll
    for (int nt = 0; nt < 4; ++nt)
#pragma unroll
      for (int e = 0; e < 8; ++e) s[nt][e] *= kscale;   // base-2 logits

    // Row max: element e lives on row e+8*half; reduce across 16 lanes of a half.
    float mx[8];
#pragma unroll
    for (int e = 0; e < 8; ++e) {
      mx[e] = fmaxf(fmaxf(s[0][e], s[1][e]), fmaxf(s[2][e], s[3][e]));
#pragma unroll
      for (int off = 1; off < 16; off <<= 1)
        mx[e] = fmaxf(mx[e], __shfl_xor(mx[e], off, 32));
    }
#pragma unroll
    for (int e = 0; e < 8; ++e) {
      const float mnew = fmaxf(mrow[e], mx[e]);
      const float fr   = exp2f(mrow[e] - mnew);
      mrow[e] = mnew;
      lrow[e] *= fr;
#pragma unroll
      for (int nt = 0; nt < 4; ++nt) o[nt][e] *= fr;
    }
    float sum[8];
#pragma unroll
    for (int e = 0; e < 8; ++e) sum[e] = 0.0f;
#pragma unroll
    for (int nt = 0; nt < 4; ++nt)
#pragma unroll
      for (int e = 0; e < 8; ++e) {
        const float p = exp2f(s[nt][e] - mrow[e]);
        sum[e] += p;
        Plds[wave][e + 8 * half][nt * 16 + r] = (_Float16)p;  // C-layout -> LDS
      }
#pragma unroll
    for (int e = 0; e < 8; ++e) {
#pragma unroll
      for (int off = 1; off < 16; off <<= 1) sum[e] += __shfl_xor(sum[e], off, 32);
      lrow[e] += sum[e];
    }
    __syncthreads();
    // O += P * V  (P re-read from LDS in A-layout; Vt rows are contiguous)
#pragma unroll
    for (int ks = 0; ks < 2; ++ks) {
      const v16h pa = frag_ld(&Plds[wave][r][ks * 32], half);
#pragma unroll
      for (int nt = 0; nt < 4; ++nt) {
        const v16h vf =
            frag_ld(Vh + (size_t)(nt * 16 + r) * SEQ + kbase + ks * 32, half);
        o[nt] = wmma16x16x32(pa, vf, o[nt]);
      }
    }
    __syncthreads();
  }

  // Normalize and write concat-heads layout AO[b][n][h*64+d] (f16).
  const int b = bh >> 4, h = bh & 15;
#pragma unroll
  for (int e = 0; e < 8; ++e) lrow[e] = 1.0f / lrow[e];
#pragma unroll
  for (int nt = 0; nt < 4; ++nt)
#pragma unroll
    for (int e = 0; e < 8; ++e) {
      const int n = q0 + e + 8 * half;
      AO[((size_t)(b * SEQ + n)) * DIM + h * DHEAD + nt * 16 + r] =
          (_Float16)(o[nt][e] * lrow[e]);
    }
}

// ---------------------------------------------------------------------------
// Kernel 3: output projection.  Out[4096,1024] = AO(f16) * Wout(f32->f16) + b
// A tile staged with gfx1250 async global->LDS DMA (no VGPR round-trip).
// ---------------------------------------------------------------------------
__global__ __launch_bounds__(256)
void out_gemm_kernel(const _Float16* __restrict__ A, const float* __restrict__ W,
                     const float* __restrict__ bias, float* __restrict__ Out) {
  __shared__ __align__(16) _Float16 As[128][32];
  __shared__ __align__(16) _Float16 Bs[128][32];
  const int tid  = threadIdx.x;
  const int m0   = blockIdx.y * 128, n0 = blockIdx.x * 128;
  const int wave = tid >> 5, lane = tid & 31;
  const int r    = lane & 15, half = lane >> 4;
  const int wm   = wave & 3,  wn   = wave >> 2;

  v8f acc[2][4];
#pragma unroll
  for (int t = 0; t < 2; ++t)
#pragma unroll
    for (int u = 0; u < 4; ++u) acc[t][u] = v8f_zero();

  for (int kt = 0; kt < DIM / 32; ++kt) {
    const int k0 = kt * 32;
#pragma unroll
    for (int i = 0; i < 2; ++i) {          // A tile: raw f16 copy, async DMA
      const int lin = tid + i * 256;
      const int row = lin >> 2, c8 = (lin & 3) * 8;
      async_copy_b128(A + (size_t)(m0 + row) * DIM + k0 + c8, &As[row][c8]);
    }
#pragma unroll
    for (int i = 0; i < 4; ++i) {          // W tile: fp32 -> f16, transposed
      const int lin  = tid + i * 256;
      const int krow = lin >> 5, c4 = (lin & 31) * 4;
      const float4 v = *reinterpret_cast<const float4*>(
          W + (size_t)(k0 + krow) * DIM + n0 + c4);
      Bs[c4 + 0][krow] = (_Float16)v.x;
      Bs[c4 + 1][krow] = (_Float16)v.y;
      Bs[c4 + 2][krow] = (_Float16)v.z;
      Bs[c4 + 3][krow] = (_Float16)v.w;
    }
    async_wait_all();                      // all async LDS writes landed
    __syncthreads();
    v16h af[2], bf[4];
#pragma unroll
    for (int t = 0; t < 2; ++t) af[t] = frag_ld(&As[wm * 32 + t * 16 + r][0], half);
#pragma unroll
    for (int u = 0; u < 4; ++u) bf[u] = frag_ld(&Bs[wn * 64 + u * 16 + r][0], half);
#pragma unroll
    for (int t = 0; t < 2; ++t)
#pragma unroll
      for (int u = 0; u < 4; ++u) acc[t][u] = wmma16x16x32(af[t], bf[u], acc[t][u]);
    __syncthreads();
  }

#pragma unroll
  for (int t = 0; t < 2; ++t)
#pragma unroll
    for (int u = 0; u < 4; ++u) {
      const int   c  = n0 + wn * 64 + u * 16 + r;
      const float bb = bias[c];
#pragma unroll
      for (int e = 0; e < 8; ++e) {
        const int m = m0 + wm * 32 + t * 16 + e + 8 * half;
        Out[(size_t)m * DIM + c] = acc[t][u][e] + bb;
      }
    }
}

// ---------------------------------------------------------------------------
extern "C" void kernel_launch(void* const* d_in, const int* in_sizes, int n_in,
                              void* d_out, int out_size, void* d_ws, size_t ws_size,
                              hipStream_t stream) {
  (void)in_sizes; (void)n_in; (void)out_size; (void)ws_size;
  const float* x     = (const float*)d_in[0];
  const float* w_qkv = (const float*)d_in[1];
  const float* w_out = (const float*)d_in[2];
  const float* b_out = (const float*)d_in[3];
  float* out = (float*)d_out;

  // Workspace layout (f16): Q | K | Vt | AO  == 4 * 8 MB = 32 MB.
  _Float16* qw = (_Float16*)d_ws;
  _Float16* kw = qw + QKV_ELEMS;
  _Float16* vt = kw + QKV_ELEMS;
  _Float16* ao = vt + QKV_ELEMS;

  qkv_gemm_kernel<<<dim3(N_QKV / 128, MROWS / 128), 256, 0, stream>>>(
      x, w_qkv, qw, kw, vt);
  attn_kernel<<<dim3(SEQ / 64, BATCH * HEADS), 128, 0, stream>>>(qw, kw, vt, ao);
  out_gemm_kernel<<<dim3(DIM / 128, MROWS / 128), 256, 0, stream>>>(
      ao, w_out, b_out, out);
}
